// ADFC_51960514347015
// MI455X (gfx1250) — compile-verified, matrix operand order checked by reading, not run on hardware
//
#include <hip/hip_runtime.h>
#include <hip/hip_bf16.h>
#include <math.h>

typedef __attribute__((ext_vector_type(16))) __bf16 v16bf;
typedef __attribute__((ext_vector_type(8)))  __bf16 v8bf;
typedef __attribute__((ext_vector_type(8)))  float  v8f;
typedef __attribute__((ext_vector_type(4)))  float  v4f;
typedef int v4i_b128 __attribute__((vector_size(4 * sizeof(int))));

#define BS   32
#define C    128
#define HH   64
#define WW   64
#define KK   4
#define HID  8
#define HW   (HH*WW)          // 4096

#if __has_builtin(__builtin_amdgcn_global_load_async_to_lds_b128)
#define HAS_ASYNC_LDS 1
#else
#define HAS_ASYNC_LDS 0
#endif

#if __has_builtin(__builtin_amdgcn_s_wait_asynccnt)
#define WAIT_ASYNC() __builtin_amdgcn_s_wait_asynccnt(0)
#else
#define WAIT_ASYNC() asm volatile("s_wait_asynccnt 0x0" ::: "memory")
#endif

#if __has_builtin(__builtin_amdgcn_sched_barrier)
#define SCHED_FENCE() __builtin_amdgcn_sched_barrier(0)
#else
#define SCHED_FENCE()
#endif

// ---------------- 1) global average pool: x[b,c,:,:] -> pooled[b,c] -----------
__global__ __launch_bounds__(256) void k_pool(const float* __restrict__ x,
                                              float* __restrict__ pooled) {
    int bc = blockIdx.x;                       // 0..4095 = b*128+c
    const float* p = x + (size_t)bc * HW;
    float s = 0.f;
    for (int i = threadIdx.x; i < HW; i += 256) s += p[i];
    __shared__ float red[256];
    red[threadIdx.x] = s;
    __syncthreads();
    for (int off = 128; off > 0; off >>= 1) {
        if (threadIdx.x < off) red[threadIdx.x] += red[threadIdx.x + off];
        __syncthreads();
    }
    if (threadIdx.x == 0) pooled[bc] = red[0] * (1.f / (float)HW);
}

// ---------------- 2) SE MLP: pooled -> att[b,k], one wave32, thread=sample ----
__device__ __forceinline__ float softplusf(float v) {
    return (v > 20.f) ? v : log1pf(expf(v));
}

__global__ void k_se(const float* __restrict__ pooled,
                     const float* __restrict__ se_w1,
                     const float* __restrict__ gn1_s, const float* __restrict__ gn1_b,
                     const float* __restrict__ se_w2,
                     const float* __restrict__ gn2_s, const float* __restrict__ gn2_b,
                     float* __restrict__ att) {
    int b = threadIdx.x;
    if (b >= BS) return;
    const float* pv = pooled + b * C;
    float h1[HID];
    for (int j = 0; j < HID; ++j) {
        float s = 0.f;
        for (int c = 0; c < C; ++c) s += pv[c] * se_w1[j * C + c];
        h1[j] = s;
    }
    // groupnorm(1, HID) + mish
    float m = 0.f;
    for (int j = 0; j < HID; ++j) m += h1[j];
    m *= (1.f / HID);
    float var = 0.f;
    for (int j = 0; j < HID; ++j) { float d = h1[j] - m; var += d * d; }
    var *= (1.f / HID);
    float inv = rsqrtf(var + 1e-5f);
    for (int j = 0; j < HID; ++j) {
        float v = (h1[j] - m) * inv * gn1_s[j] + gn1_b[j];
        h1[j] = v * tanhf(softplusf(v));
    }
    float h2[KK];
    for (int k = 0; k < KK; ++k) {
        float s = 0.f;
        for (int j = 0; j < HID; ++j) s += h1[j] * se_w2[k * HID + j];
        h2[k] = s;
    }
    m = 0.f;
    for (int k = 0; k < KK; ++k) m += h2[k];
    m *= (1.f / KK);
    var = 0.f;
    for (int k = 0; k < KK; ++k) { float d = h2[k] - m; var += d * d; }
    var *= (1.f / KK);
    inv = rsqrtf(var + 1e-5f);
    for (int k = 0; k < KK; ++k) {
        float v = (h2[k] - m) * inv * gn2_s[k] + gn2_b[k];
        att[b * KK + k] = 1.f / (1.f + expf(-v));
    }
}

// ---------------- 3a) weight aggregation -> bf16, layout [b][tap][co][ci] -----
__global__ __launch_bounds__(256) void k_aggw(const float* __restrict__ weight,
                                              const float* __restrict__ att,
                                              __bf16* __restrict__ aggW) {
    size_t i = (size_t)blockIdx.x * 256 + threadIdx.x;  // total 32*9*128*128
    int ci  = (int)(i & 127);
    size_t t = i >> 7;
    int co  = (int)(t & 127);  t >>= 7;
    int tap = (int)(t % 9);
    int b   = (int)(t / 9);
    float a0 = att[b * 4 + 0], a1 = att[b * 4 + 1],
          a2 = att[b * 4 + 2], a3 = att[b * 4 + 3];
    size_t wb = (size_t)(co * C + ci) * 9 + tap;        // k-stride = 128*128*9
    const size_t KS = (size_t)C * C * 9;
    float s = a0 * weight[wb] + a1 * weight[wb + KS]
            + a2 * weight[wb + 2 * KS] + a3 * weight[wb + 3 * KS];
    aggW[i] = (__bf16)s;
}

// ---------------- 3b) bias aggregation: aggb[b,co] ----------------------------
__global__ __launch_bounds__(256) void k_aggb(const float* __restrict__ bias,
                                              const float* __restrict__ att,
                                              float* __restrict__ aggb) {
    int i  = blockIdx.x * 256 + threadIdx.x;  // 4096
    int co = i & 127, b = i >> 7;
    float s = 0.f;
    for (int k = 0; k < KK; ++k) s += att[b * 4 + k] * bias[k * C + co];
    aggb[i] = s;
}

// ---------------- 4) implicit-GEMM conv via v_wmma_f32_16x16x32_bf16 ----------
// Workgroup = 8 wave32 (256 thr). Tile: M=128 spatial (2 rows x 64 cols),
// N=128 out-ch, K = 4 ci-chunks x 9 taps x 32.
// B tiles double-buffered in LDS; the async global->LDS copy for step s+1 is
// issued before computing step s, so the copy hides behind the WMMA chain.
__global__ __launch_bounds__(256) void k_conv(const float* __restrict__ x,
                                              const __bf16* __restrict__ aggW,
                                              const float* __restrict__ aggb,
                                              float* __restrict__ out) {
    // xs: bf16 x halo, [row(4)][col(66)][ch(32)] -> ch contiguous for ds_read_b128
    __shared__ __align__(16) __bf16 xs[4 * 66 * 32];
    // bsh: B tile double buffer, [buf(2)][n(128)][kk(32)]
    __shared__ __align__(16) __bf16 bsh[2][128 * 32];

    int b   = blockIdx.x >> 5;          // sample
    int h0  = (blockIdx.x & 31) * 2;    // first output row of tile
    int tid = threadIdx.x;
    int lane = tid & 31;
    int wv   = tid >> 5;                // wave id 0..7: M-subtile
    int l15  = lane & 15;
    int khalf = (lane >> 4) * 8;        // K-half select per ISA A/B layout

    const float*  xb = x    + (size_t)b * C * HW;
    const __bf16* wb = aggW + (size_t)b * 9 * C * C;

    // stage one 8KB B tile (tap, ci0) into bsh[bufn]; 512 x 16B chunks, 2/thread
    auto stageB = [&](int ci0n, int tapn, int bufn) {
        const __bf16* wt = wb + (size_t)tapn * C * C + ci0n;
#if HAS_ASYNC_LDS
        #pragma unroll
        for (int r = 0; r < 2; ++r) {
            int c2 = tid + r * 256;
            int kb = c2 & 3;             // kk block of 8
            int n  = c2 >> 2;            // out channel
            __builtin_amdgcn_global_load_async_to_lds_b128(
                (__attribute__((address_space(1))) v4i_b128*)(wt + n * C + kb * 8),
                (__attribute__((address_space(3))) v4i_b128*)&bsh[bufn][n * 32 + kb * 8],
                0, 0);
        }
#else
        #pragma unroll
        for (int r = 0; r < 2; ++r) {
            int c2 = tid + r * 256;
            int kb = c2 & 3;
            int n  = c2 >> 2;
            *(v8bf*)&bsh[bufn][n * 32 + kb * 8] = *(const v8bf*)(wt + n * C + kb * 8);
        }
#endif
    };

    // zero the always-out-of-bounds edge columns (col 0 and 65) exactly once:
    // 4 rows x 2 cols x 32 ch = 256 elements, one per thread
    {
        int row  = tid >> 6;
        int edge = (tid >> 5) & 1;
        int ch   = tid & 31;
        xs[(row * 66 + (edge ? 65 : 0)) * 32 + ch] = (__bf16)0.f;
    }

    v8f zero = {0.f, 0.f, 0.f, 0.f, 0.f, 0.f, 0.f, 0.f};
    v8f acc[8];
    #pragma unroll
    for (int nt = 0; nt < 8; ++nt) acc[nt] = zero;

    // per-thread x staging geometry: one (row, col) per thread, 32 channels each
    int srow = tid >> 6;                 // 0..3
    int scol = (tid & 63) + 1;           // 1..64  (ww = scol-1 in 0..63, always valid)
    int shh  = h0 - 1 + srow;
    bool sok = (unsigned)shh < (unsigned)HH;
    int shcl = sok ? shh : 0;            // clamped safe row
    const float* sgp0 = xb + (size_t)shcl * WW + (scol - 1);
    __bf16* slp = &xs[(srow * 66 + scol) * 32];

    int step = 0;
    for (int ci0 = 0; ci0 < C; ci0 += 32) {
        // xs WAR safe: end-of-step barrier followed all register-landed reads.
        // branch-free staging: 4 groups of 8 independent loads -> 1 ds_store_b128
        {
            const float* sgp = sgp0 + (size_t)ci0 * HW;
            #pragma unroll
            for (int cg = 0; cg < 4; ++cg) {
                float f[8];
                #pragma unroll
                for (int u = 0; u < 8; ++u)
                    f[u] = sgp[(size_t)(cg * 8 + u) * HW];
                v8bf pk;
                #pragma unroll
                for (int u = 0; u < 8; ++u)
                    pk[u] = (__bf16)(sok ? f[u] : 0.f);
                *(v8bf*)(slp + cg * 8) = pk;
            }
        }
        if (step == 0) {                 // prologue: stage B for the very first step
            stageB(0, 0, 0);
            WAIT_ASYNC();
        }
        __syncthreads();                 // publish xs (+ bsh[0] on first chunk)

        for (int tap = 0; tap < 9; ++tap, ++step) {
            int kh = tap / 3, kw = tap % 3;
            int buf = step & 1;

            // prefetch next step's B tile into the other buffer (hidden by WMMAs)
            if (step + 1 < 36) {
                int nci0 = (tap < 8) ? ci0 : ci0 + 32;
                int ntap = (tap < 8) ? tap + 1 : 0;
                stageB(nci0, ntap, (step + 1) & 1);
            }

            // A fragment: lane row m = l15 (col = wavebase + l15 + kw), K striped per ISA
            const __bf16* ap = &xs[(((wv >> 2) + kh) * 66 + (wv & 3) * 16 + l15 + kw) * 32];
            v8bf alo = *(const v8bf*)(ap + khalf);
            v8bf ahi = *(const v8bf*)(ap + 16 + khalf);
            v16bf afrag = __builtin_shufflevector(alo, ahi,
                0, 1, 2, 3, 4, 5, 6, 7, 8, 9, 10, 11, 12, 13, 14, 15);

            // preload all 8 B fragments so ds_loads pipeline...
            v8bf blo[8], bhi[8];
            #pragma unroll
            for (int nt = 0; nt < 8; ++nt) {
                const __bf16* bp = &bsh[buf][(nt * 16 + l15) * 32];
                blo[nt] = *(const v8bf*)(bp + khalf);
                bhi[nt] = *(const v8bf*)(bp + 16 + khalf);
            }
            SCHED_FENCE();   // keep the 16 ds_loads ahead of the WMMA chain
            // ...then WMMA back-to-back
            #pragma unroll
            for (int nt = 0; nt < 8; ++nt) {
                v16bf bfrag = __builtin_shufflevector(blo[nt], bhi[nt],
                    0, 1, 2, 3, 4, 5, 6, 7, 8, 9, 10, 11, 12, 13, 14, 15);
                acc[nt] = __builtin_amdgcn_wmma_f32_16x16x32_bf16(
                    false, afrag, false, bfrag, (short)0, acc[nt], false, false);
            }

#if HAS_ASYNC_LDS
            WAIT_ASYNC();        // this wave's prefetch for step+1 has landed in LDS
#endif
            __syncthreads();     // all waves done with step; publish bsh[(step+1)&1]
        }
    }

    // epilogue: D layout -> lane l15 = N, VGPR r -> M = r + 8*(lane>>4)
    int posBase = wv * 16 + 8 * (lane >> 4);   // multiple of 8: run of 8 stays in one row
    int h  = h0 + (posBase >> 6);
    int wc = posBase & 63;
    #pragma unroll
    for (int nt = 0; nt < 8; ++nt) {
        int co = nt * 16 + l15;
        float ab = aggb[b * C + co];
        float* dst = out + (((size_t)(b * C + co) * HH + h) * WW + wc);
        v4f lo = {acc[nt][0] + ab, acc[nt][1] + ab, acc[nt][2] + ab, acc[nt][3] + ab};
        v4f hi = {acc[nt][4] + ab, acc[nt][5] + ab, acc[nt][6] + ab, acc[nt][7] + ab};
        *(v4f*)dst       = lo;
        *(v4f*)(dst + 4) = hi;
    }
}

extern "C" void kernel_launch(void* const* d_in, const int* in_sizes, int n_in,
                              void* d_out, int out_size, void* d_ws, size_t ws_size,
                              hipStream_t stream) {
    const float* x      = (const float*)d_in[0];
    const float* weight = (const float*)d_in[1];
    const float* bias   = (const float*)d_in[2];
    const float* se_w1  = (const float*)d_in[3];
    const float* gn1_s  = (const float*)d_in[4];
    const float* gn1_b  = (const float*)d_in[5];
    const float* se_w2  = (const float*)d_in[6];
    const float* gn2_s  = (const float*)d_in[7];
    const float* gn2_b  = (const float*)d_in[8];
    float* out = (float*)d_out;

    // workspace layout
    char* ws = (char*)d_ws;
    float*  pooled = (float*)(ws);                       // 32*128 f32   = 16 KB
    float*  att    = (float*)(ws + 16384);               // 32*4   f32
    float*  aggb   = (float*)(ws + 16384 + 512);         // 32*128 f32   = 16 KB
    __bf16* aggW   = (__bf16*)(ws + 33280);              // 32*9*128*128 bf16 = 9.44 MB

    k_pool<<<BS * C, 256, 0, stream>>>(x, pooled);
    k_se<<<1, 32, 0, stream>>>(pooled, se_w1, gn1_s, gn1_b, se_w2, gn2_s, gn2_b, att);
    {
        size_t total = (size_t)BS * 9 * C * C;           // 4,718,592
        k_aggw<<<(int)(total / 256), 256, 0, stream>>>(weight, att, aggW);
    }
    k_aggb<<<(BS * C) / 256, 256, 0, stream>>>(bias, att, aggb);
    k_conv<<<BS * (HH / 2), 256, 0, stream>>>(x, aggW, aggb, out);
}
